// MCCDecoderAttention_10892037063169
// MI455X (gfx1250) — matrix operand order repaired and checked by reading, not compiled, not run on hardware
//
#include <hip/hip_runtime.h>
#include <stdint.h>

#define NTOK   3072
#define DIM    768
#define HEADS  12
#define HD     64
#define QLEN   550
#define NOBJ   8
#define MPATCH (NTOK - NOBJ - QLEN)   /* 2514 */
#define CTX    (NTOK - QLEN)          /* 2522 */
#define ATTN_SCALE 0.125f             /* 64^-0.5 */
#define LDK    48                     /* padded LDS row stride (bf16 elems), 96B rows */

typedef __attribute__((ext_vector_type(16))) __bf16 bf16x16;
typedef __attribute__((ext_vector_type(8)))  float  floatx8;

union FragU { uint32_t u[8]; bf16x16 v; };

static __device__ __forceinline__ uint16_t f2bf(float x) {
    uint32_t u = __builtin_bit_cast(uint32_t, x);
    uint32_t r = u + 0x7fffu + ((u >> 16) & 1u);   // round-to-nearest-even
    return (uint16_t)(r >> 16);
}

// A-matrix fragment (16x32 bf16) from a row-major [M][K] tile.
// lane m = lane&15; lanes 0-15: K {0..7,16..23}; lanes 16-31: K {8..15,24..31}.
static __device__ __forceinline__ bf16x16 load_frag_a(const uint16_t* base, int ldk, int lane) {
    const int m  = lane & 15;
    const int kb = (lane & 16) ? 8 : 0;
    const uint32_t* p0 = (const uint32_t*)(base + m * ldk + kb);
    const uint32_t* p1 = (const uint32_t*)(base + m * ldk + kb + 16);
    FragU f;
#pragma unroll
    for (int i = 0; i < 4; ++i) { f.u[i] = p0[i]; f.u[4 + i] = p1[i]; }
    return f.v;
}

// B-matrix fragment (32x16 bf16) from a row-major [N][K] tile.
// lane n = lane&15; lanes 0-15: K {0..7,16..23}; lanes 16-31: K {8..15,24..31}.
static __device__ __forceinline__ bf16x16 load_frag_b(const uint16_t* base, int ldk, int lane) {
    const int n  = lane & 15;
    const int kb = (lane & 16) ? 8 : 0;
    const uint32_t* p = (const uint32_t*)(base + n * ldk + kb);
    const uint32_t* p1 = (const uint32_t*)(base + n * ldk + kb + 16);
    FragU f;
#pragma unroll
    for (int i = 0; i < 4; ++i) { f.u[i] = p[i]; f.u[4 + i] = p1[i]; }
    return f.v;
}

static __device__ __forceinline__ floatx8 wmma_bf16(bf16x16 a, bf16x16 b, floatx8 c) {
    return __builtin_amdgcn_wmma_f32_16x16x32_bf16(false, a, false, b, (short)0, c, false, false);
}

// ---------------------------------------------------------------------------
// Kernel 0: segment min/max of patch positions -> lo/hi per object (token cols)
// ---------------------------------------------------------------------------
__global__ void mcc_prep_lohi(const int* __restrict__ patch_idx, int* __restrict__ lohi) {
    int tid = threadIdx.x;
    if (tid < NOBJ) { lohi[tid] = 0x7fffffff; lohi[NOBJ + tid] = (int)0x80000000; }
    __syncthreads();
    for (int i = tid; i < MPATCH; i += blockDim.x) {
        int o = patch_idx[i];
        atomicMin(&lohi[o], i);
        atomicMax(&lohi[NOBJ + o], i);
    }
    __syncthreads();
    if (tid < NOBJ) {
        int lo = lohi[tid], hi = lohi[NOBJ + tid];
        lohi[tid]        = (lo == 0x7fffffff)      ? 0x3fffffff  : lo + NOBJ;
        lohi[NOBJ + tid] = (hi == (int)0x80000000) ? -0x3fffffff : hi + NOBJ;
    }
}

// ---------------------------------------------------------------------------
// Kernel 1: QKV GEMM  C[3072,2304] = x @ Wqkv^T, scattered to q/k row-major
// bf16 and v transposed bf16.  Block tile 128x64, K-step 32, 8 waves (4x2),
// wave tile 32x32 = 2x2 WMMA accumulators.
// ---------------------------------------------------------------------------
__global__ __launch_bounds__(256) void mcc_qkv_gemm(const float* __restrict__ x,
                                                    const float* __restrict__ wqkv,
                                                    uint16_t* __restrict__ qb,
                                                    uint16_t* __restrict__ kb,
                                                    uint16_t* __restrict__ vt) {
    __shared__ uint16_t As[128 * LDK];
    __shared__ uint16_t Bs[64 * LDK];
    const int tid  = threadIdx.x;
    const int lane = tid & 31;
    const int wave = tid >> 5;
    const int wm = wave >> 1, wn = wave & 1;
    const int m0 = blockIdx.x * 128;
    const int n0 = blockIdx.y * 64;

    floatx8 z = {};
    floatx8 acc[2][2];
#pragma unroll
    for (int i = 0; i < 2; ++i)
#pragma unroll
        for (int j = 0; j < 2; ++j) acc[i][j] = z;

    for (int k0 = 0; k0 < DIM; k0 += 32) {
        // stage A tile 128x32 f32 -> bf16
        for (int s = tid; s < 1024; s += 256) {
            int row = s >> 3, c4 = s & 7;
            float4 f = *(const float4*)(x + (size_t)(m0 + row) * DIM + k0 + c4 * 4);
            uint32_t lo = (uint32_t)f2bf(f.x) | ((uint32_t)f2bf(f.y) << 16);
            uint32_t hi = (uint32_t)f2bf(f.z) | ((uint32_t)f2bf(f.w) << 16);
            uint32_t* d = (uint32_t*)(&As[row * LDK + c4 * 4]);
            d[0] = lo; d[1] = hi;
        }
        // stage B tile 64x32 f32 -> bf16 (Wqkv rows are K-contiguous already)
        for (int s = tid; s < 512; s += 256) {
            int row = s >> 3, c4 = s & 7;
            float4 f = *(const float4*)(wqkv + (size_t)(n0 + row) * DIM + k0 + c4 * 4);
            uint32_t lo = (uint32_t)f2bf(f.x) | ((uint32_t)f2bf(f.y) << 16);
            uint32_t hi = (uint32_t)f2bf(f.z) | ((uint32_t)f2bf(f.w) << 16);
            uint32_t* d = (uint32_t*)(&Bs[row * LDK + c4 * 4]);
            d[0] = lo; d[1] = hi;
        }
        if (k0 + 32 < DIM) {
            __builtin_prefetch(x + (size_t)(m0 + (tid >> 1)) * DIM + k0 + 32, 0, 1);
            __builtin_prefetch(wqkv + (size_t)(n0 + (tid & 63)) * DIM + k0 + 32, 0, 1);
        }
        __syncthreads();

        bf16x16 af[2], bf[2];
#pragma unroll
        for (int i = 0; i < 2; ++i) af[i] = load_frag_a(As + (wm * 32 + i * 16) * LDK, LDK, lane);
#pragma unroll
        for (int j = 0; j < 2; ++j) bf[j] = load_frag_b(Bs + (wn * 32 + j * 16) * LDK, LDK, lane);
#pragma unroll
        for (int i = 0; i < 2; ++i)
#pragma unroll
            for (int j = 0; j < 2; ++j) acc[i][j] = wmma_bf16(af[i], bf[j], acc[i][j]);
        __syncthreads();
    }

    // epilogue: scatter to q / k (row-major [head][tok][64]) or vT ([head][64][tok])
    const int part  = n0 / DIM;          // 0=q 1=k 2=v (uniform per block)
    const int head  = (n0 % DIM) / HD;   // uniform per block
    const int mhalf = (lane & 16) ? 8 : 0;
    const int nn    = lane & 15;
#pragma unroll
    for (int i = 0; i < 2; ++i)
#pragma unroll
        for (int j = 0; j < 2; ++j)
#pragma unroll
            for (int v = 0; v < 8; ++v) {
                int row = m0 + wm * 32 + i * 16 + mhalf + v;
                int d   = wn * 32 + j * 16 + nn;
                uint16_t h = f2bf(acc[i][j][v]);
                if (part == 0)      qb[((head * NTOK + row) << 6) + d] = h;
                else if (part == 1) kb[((head * NTOK + row) << 6) + d] = h;
                else                vt[(head * HD + d) * NTOK + row]   = h;
            }
}

// ---------------------------------------------------------------------------
// Kernel 2: fused masked flash attention.  One wave = 16 query rows, loop over
// 32-column chunks.  Q/K/V fragments loaded directly from global (layouts
// match WMMA frag shapes); only P goes through LDS (C-layout -> A-layout),
// synchronized with the wave-private split counter s_wait_dscnt.
// ---------------------------------------------------------------------------
__global__ __launch_bounds__(256) void mcc_attn(const uint16_t* __restrict__ qb,
                                                const uint16_t* __restrict__ kb,
                                                const uint16_t* __restrict__ vt,
                                                const int* __restrict__ point_idx,
                                                const int* __restrict__ lohi,
                                                uint16_t* __restrict__ attn_out) {
    __shared__ uint16_t Ps[8][16 * LDK];
    const int tid  = threadIdx.x;
    const int lane = tid & 31;
    const int wave = tid >> 5;
    const int head = blockIdx.y;
    const int r0   = blockIdx.x * 128 + wave * 16;
    const int mhalf = (lane & 16) ? 8 : 0;
    const int nn    = lane & 15;
    const float NEG = -__builtin_inff();

    const uint16_t* qh = qb + (size_t)head * NTOK * HD;
    const uint16_t* kh = kb + (size_t)head * NTOK * HD;
    const uint16_t* vh = vt + (size_t)head * HD * NTOK;

    // Q fragments for this wave's 16 rows (head_dim 64 = 2 K-steps)
    bf16x16 aQ[2];
#pragma unroll
    for (int t = 0; t < 2; ++t) aQ[t] = load_frag_a(qh + r0 * HD + t * 32, HD, lane);

    // per-row mask metadata (row M = mhalf + v for this lane)
    int rv[8], pid[8], rlo[8], rhi[8];
    bool isq[8];
#pragma unroll
    for (int v = 0; v < 8; ++v) {
        rv[v]  = r0 + mhalf + v;
        isq[v] = rv[v] >= CTX;
        if (isq[v]) {
            pid[v] = point_idx[rv[v] - CTX];
            rlo[v] = lohi[pid[v]];
            rhi[v] = lohi[NOBJ + pid[v]];
        } else { pid[v] = -1; rlo[v] = 0x3fffffff; rhi[v] = -1; }
    }

    floatx8 z = {};
    floatx8 accO[4];
#pragma unroll
    for (int j = 0; j < 4; ++j) accO[j] = z;
    float mrun[8], lrun[8];
#pragma unroll
    for (int v = 0; v < 8; ++v) { mrun[v] = NEG; lrun[v] = 0.0f; }

    // pure-context row tiles never see unmasked columns >= CTX
    const int nchunks = (r0 + 16 <= CTX) ? ((CTX + 31) / 32) : (NTOK / 32);

    for (int cc = 0; cc < nchunks; ++cc) {
        const int c0 = cc * 32;
        // S = q @ k^T for 16x32 score tile (two 16x16 C tiles)
        floatx8 s0 = z, s1 = z;
#pragma unroll
        for (int t = 0; t < 2; ++t) {
            bf16x16 b0 = load_frag_b(kh + (size_t)c0 * HD + t * 32, HD, lane);
            s0 = wmma_bf16(aQ[t], b0, s0);
            bf16x16 b1 = load_frag_b(kh + (size_t)(c0 + 16) * HD + t * 32, HD, lane);
            s1 = wmma_bf16(aQ[t], b1, s1);
        }

        float alpha[8];
#pragma unroll
        for (int v = 0; v < 8; ++v) {
            const int ca = c0 + nn, cb = c0 + 16 + nn;
            float x0 = s0[v] * ATTN_SCALE;
            float x1 = s1[v] * ATTN_SCALE;
            bool a0, a1;
            if (!isq[v]) { a0 = ca < CTX; a1 = cb < CTX; }
            else {
                a0 = (ca == pid[v]) || (ca >= rlo[v] && ca <= rhi[v]) || (ca == rv[v]);
                a1 = (cb == pid[v]) || (cb >= rlo[v] && cb <= rhi[v]) || (cb == rv[v]);
            }
            x0 = a0 ? x0 : NEG;
            x1 = a1 ? x1 : NEG;
            // row max across the 16 lanes sharing this row
            float rm = fmaxf(x0, x1);
#pragma unroll
            for (int msk = 1; msk < 16; msk <<= 1) rm = fmaxf(rm, __shfl_xor(rm, msk, 32));
            float mn = fmaxf(mrun[v], rm);            // finite after chunk 0 for every row
            alpha[v] = __expf(mrun[v] - mn);
            mrun[v]  = mn;
            float p0 = __expf(x0 - mn);
            float p1 = __expf(x1 - mn);
            float rs = p0 + p1;
#pragma unroll
            for (int msk = 1; msk < 16; msk <<= 1) rs += __shfl_xor(rs, msk, 32);
            lrun[v] = lrun[v] * alpha[v] + rs;
            // stage P in A-matrix source layout: [16 M][32 K] row-major bf16
            Ps[wave][(mhalf + v) * LDK + nn]      = f2bf(p0);
            Ps[wave][(mhalf + v) * LDK + 16 + nn] = f2bf(p1);
        }
        asm volatile("s_wait_dscnt 0" ::: "memory");   // wave-private LDS RAW
        bf16x16 aP = load_frag_a(&Ps[wave][0], LDK, lane);
#pragma unroll
        for (int j = 0; j < 4; ++j) {
#pragma unroll
            for (int v = 0; v < 8; ++v) accO[j][v] *= alpha[v];
            bf16x16 bV = load_frag_b(vh + (size_t)(j * 16) * NTOK + c0, NTOK, lane);
            accO[j] = wmma_bf16(aP, bV, accO[j]);
        }
    }

    // normalize and emit bf16 [3072, 768] attention output
#pragma unroll
    for (int j = 0; j < 4; ++j)
#pragma unroll
        for (int v = 0; v < 8; ++v) {
            int row = r0 + mhalf + v;
            int d   = j * 16 + nn;
            attn_out[(size_t)row * DIM + head * HD + d] = f2bf(accO[j][v] / lrun[v]);
        }
}

// ---------------------------------------------------------------------------
// Kernel 3: output projection  out = attn(bf16) @ Wproj^T + bproj  (f32 out)
// ---------------------------------------------------------------------------
__global__ __launch_bounds__(256) void mcc_proj_gemm(const uint16_t* __restrict__ a,
                                                     const float* __restrict__ wproj,
                                                     const float* __restrict__ bproj,
                                                     float* __restrict__ out) {
    __shared__ uint16_t As[128 * LDK];
    __shared__ uint16_t Bs[64 * LDK];
    const int tid  = threadIdx.x;
    const int lane = tid & 31;
    const int wave = tid >> 5;
    const int wm = wave >> 1, wn = wave & 1;
    const int m0 = blockIdx.x * 128;
    const int n0 = blockIdx.y * 64;

    floatx8 z = {};
    floatx8 acc[2][2];
#pragma unroll
    for (int i = 0; i < 2; ++i)
#pragma unroll
        for (int j = 0; j < 2; ++j) acc[i][j] = z;

    for (int k0 = 0; k0 < DIM; k0 += 32) {
        // A already bf16: 128x32 = 512 x uint4 (8 bf16) slots
        for (int s = tid; s < 512; s += 256) {
            int row = s >> 2, c8 = s & 3;
            uint4 q = *(const uint4*)(a + (size_t)(m0 + row) * DIM + k0 + c8 * 8);
            *(uint4*)(&As[row * LDK + c8 * 8]) = q;
        }
        for (int s = tid; s < 512; s += 256) {
            int row = s >> 3, c4 = s & 7;
            float4 f = *(const float4*)(wproj + (size_t)(n0 + row) * DIM + k0 + c4 * 4);
            uint32_t lo = (uint32_t)f2bf(f.x) | ((uint32_t)f2bf(f.y) << 16);
            uint32_t hi = (uint32_t)f2bf(f.z) | ((uint32_t)f2bf(f.w) << 16);
            uint32_t* d = (uint32_t*)(&Bs[row * LDK + c4 * 4]);
            d[0] = lo; d[1] = hi;
        }
        if (k0 + 32 < DIM) {
            __builtin_prefetch(a + (size_t)(m0 + (tid >> 1)) * DIM + k0 + 32, 0, 1);
            __builtin_prefetch(wproj + (size_t)(n0 + (tid & 63)) * DIM + k0 + 32, 0, 1);
        }
        __syncthreads();

        bf16x16 af[2], bf[2];
#pragma unroll
        for (int i = 0; i < 2; ++i) af[i] = load_frag_a(As + (wm * 32 + i * 16) * LDK, LDK, lane);
#pragma unroll
        for (int j = 0; j < 2; ++j) bf[j] = load_frag_b(Bs + (wn * 32 + j * 16) * LDK, LDK, lane);
#pragma unroll
        for (int i = 0; i < 2; ++i)
#pragma unroll
            for (int j = 0; j < 2; ++j) acc[i][j] = wmma_bf16(af[i], bf[j], acc[i][j]);
        __syncthreads();
    }

    const int mhalf = (lane & 16) ? 8 : 0;
    const int nn    = lane & 15;
#pragma unroll
    for (int i = 0; i < 2; ++i)
#pragma unroll
        for (int j = 0; j < 2; ++j)
#pragma unroll
            for (int v = 0; v < 8; ++v) {
                int row = m0 + wm * 32 + i * 16 + mhalf + v;
                int col = n0 + wn * 32 + j * 16 + nn;
                out[(size_t)row * DIM + col] = acc[i][j][v] + bproj[col];
            }
}

// ---------------------------------------------------------------------------
extern "C" void kernel_launch(void* const* d_in, const int* in_sizes, int n_in,
                              void* d_out, int out_size, void* d_ws, size_t ws_size,
                              hipStream_t stream) {
    (void)in_sizes; (void)n_in; (void)out_size; (void)ws_size;
    const float* x         = (const float*)d_in[0];
    const int*   patch_idx = (const int*)d_in[1];
    const int*   point_idx = (const int*)d_in[2];
    /* d_in[3] = n_patches scalar (== NOBJ) */
    const float* wqkv      = (const float*)d_in[4];
    const float* wproj     = (const float*)d_in[5];
    const float* bproj     = (const float*)d_in[6];
    float*       out       = (float*)d_out;

    char* w = (char*)d_ws;
    int* lohi = (int*)w;                         w += 256;
    uint16_t* qb   = (uint16_t*)w;               w += (size_t)HEADS * NTOK * HD * 2;
    uint16_t* kb   = (uint16_t*)w;               w += (size_t)HEADS * NTOK * HD * 2;
    uint16_t* vt   = (uint16_t*)w;               w += (size_t)HEADS * NTOK * HD * 2;
    uint16_t* attn = (uint16_t*)w;               w += (size_t)NTOK * DIM * 2;

    mcc_prep_lohi<<<1, 256, 0, stream>>>(patch_idx, lohi);
    mcc_qkv_gemm<<<dim3(NTOK / 128, (3 * DIM) / 64), 256, 0, stream>>>(x, wqkv, qb, kb, vt);
    mcc_attn<<<dim3(NTOK / 128, HEADS), 256, 0, stream>>>(qb, kb, vt, point_idx, lohi, attn);
    mcc_proj_gemm<<<dim3(NTOK / 128, DIM / 64), 256, 0, stream>>>(attn, wproj, bproj, out);
}
